// SpatialAttentionLayer_10780367913076
// MI455X (gfx1250) — matrix operand order repaired and checked by reading, not compiled, not run on hardware
//
#include <hip/hip_runtime.h>
#include <hip/hip_bf16.h>
#include <math.h>

#define BS      256
#define GRID_N  49
#define LSTM_HD 512
#define CNN_C   2048
#define PROJ    512
#define M_ROWS  (BS * GRID_N)   // 12544 = 784 * 16 exactly

typedef __bf16 bf16_t;
typedef bf16_t v16bf __attribute__((ext_vector_type(16)));
typedef float  v8f   __attribute__((ext_vector_type(8)));

union BF16Vec { unsigned int u[8]; uint4 q[2]; v16bf v; };

struct ATile { uint4 r[4]; };   // 16 raw fp32 (one lane's A fragment, pre-convert)
struct BTile { uint4 q[8]; };   // 4 subtiles x 16 bf16 (one lane's B fragments)

// One-instruction fp32x2 -> bf16x2 pack (truncation) via v_perm_b32.
__device__ __forceinline__ unsigned int pack2_bf16_trunc(unsigned int lo, unsigned int hi) {
    return __builtin_amdgcn_perm(hi, lo, 0x07060302u);
}

// RNE fp32x2 -> bf16x2 pack (one-shot weight converter only)
__device__ __forceinline__ unsigned int pack2_bf16_rne(float lo, float hi) {
    unsigned int a = __float_as_uint(lo);
    unsigned int b = __float_as_uint(hi);
    a += 0x7FFFu + ((a >> 16) & 1u);
    b += 0x7FFFu + ((b >> 16) & 1u);
    return (a >> 16) | (b & 0xFFFF0000u);
}

// One-shot converter: fp32 array -> bf16 array (RNE), 4 elements/thread, grid-stride.
__global__ void k_f32_to_bf16(const float* __restrict__ src,
                              unsigned int* __restrict__ dst, int n4) {
    int i = blockIdx.x * blockDim.x + threadIdx.x;
    const int stride = gridDim.x * blockDim.x;
    for (; i < n4; i += stride) {
        const float4 f = ((const float4*)src)[i];
        uint2 o;
        o.x = pack2_bf16_rne(f.x, f.y);
        o.y = pack2_bf16_rne(f.z, f.w);
        ((uint2*)dst)[i] = o;
    }
}

// Per-lane A fragment: 16 fp32 = two contiguous 8-float chunks at +0 and +16.
__device__ __forceinline__ ATile load_a(const float* p) {
    ATile t;
    t.r[0] = *(const uint4*)(p + 0);
    t.r[1] = *(const uint4*)(p + 4);
    t.r[2] = *(const uint4*)(p + 16);
    t.r[3] = *(const uint4*)(p + 20);
    return t;
}

// Per-lane B fragments for 4 N-subtiles: 16 bf16 each = two 8-bf16 chunks at +0 and +16.
template<int KTOT>
__device__ __forceinline__ BTile load_b(const bf16_t* p) {
    BTile t;
#pragma unroll
    for (int s = 0; s < 4; ++s) {
        const uint4* bp = (const uint4*)(p + (size_t)(s * 16) * KTOT);
        t.q[s * 2 + 0] = bp[0];
        t.q[s * 2 + 1] = bp[2];
    }
    return t;
}

// Convert raw fp32 A fragment to bf16 WMMA layout (8 v_perm).
__device__ __forceinline__ void cvt_a(const ATile& t, BF16Vec& a) {
    const unsigned int* u = (const unsigned int*)&t;
#pragma unroll
    for (int i = 0; i < 8; ++i)
        a.u[i] = pack2_bf16_trunc(u[2 * i], u[2 * i + 1]);
}

__device__ __forceinline__ void compute4(const ATile& ar, const BTile& bt, v8f acc[4]) {
    BF16Vec a;
    cvt_a(ar, a);
#pragma unroll
    for (int s = 0; s < 4; ++s) {
        BF16Vec bb;
        bb.q[0] = bt.q[s * 2 + 0];
        bb.q[1] = bt.q[s * 2 + 1];
        acc[s] = __builtin_amdgcn_wmma_f32_16x16x32_bf16(
            false, a.v, false, bb.v, (short)0, acc[s], false, false);
    }
}

// Barrier-free, double-buffered bf16 WMMA GEMM: C(16 x 512) += A(16 x KTOT) * B^T.
// A: fp32 (row stride LDA), per-lane direct loads in the ISA 16-bit A layout:
//    lane<16: row=lane,    k = {0-7, 16-23}  (+kb)
//    lane>=16: row=lane-16, k = {8-15, 24-31} (+kb)
// B: pre-converted bf16 (row stride KTOT); lane = column, lanes 16-31 take K+8.
// 8 waves/block; wave w owns N columns [w*64, w*64+64) as 4 subtiles of 16.
template<int KTOT, int LDA>
__device__ __forceinline__ void gemm16xN(const float* __restrict__ A,
                                         const bf16_t* __restrict__ B,
                                         int mbase, v8f acc[4]) {
    const int lane = threadIdx.x & 31;
    const int wv   = threadIdx.x >> 5;
    const int half = lane >> 4;

    const float*  arow  = A + (size_t)(mbase + (lane & 15)) * LDA + (half ? 8 : 0);
    const bf16_t* bbase = B + (size_t)(wv * 64 + (lane & 15)) * KTOT + (half ? 8 : 0);

    // Prologue: first tile in flight.
    ATile ar = load_a(arow);
    BTile bt = load_b<KTOT>(bbase);

    // Steady state: issue next tile's 12 loads, then 4 WMMAs on current tile.
    for (int kb = 0; kb < KTOT - 32; kb += 32) {
        ATile ar_n = load_a(arow + kb + 32);
        BTile bt_n = load_b<KTOT>(bbase + kb + 32);
        compute4(ar, bt, acc);
        ar = ar_n;
        bt = bt_n;
    }
    compute4(ar, bt, acc);   // epilogue tile
}

// Kernel: H = lstm_hidden @ W_lstm^T + b_lstm   (256 x 512, fp32 out to workspace)
__global__ __launch_bounds__(256) void k_proj_lstm(const float* __restrict__ X,
                                                   const bf16_t* __restrict__ Wbf,
                                                   const float* __restrict__ bias,
                                                   float* __restrict__ Hout) {
    v8f acc[4];
#pragma unroll
    for (int s = 0; s < 4; ++s)
#pragma unroll
        for (int j = 0; j < 8; ++j) acc[s][j] = 0.0f;

    const int mbase = blockIdx.x * 16;
    gemm16xN<LSTM_HD, LSTM_HD>(X, Wbf, mbase, acc);

    const int lane = threadIdx.x & 31;
    const int wv   = threadIdx.x >> 5;
    const int rofs = (lane >> 4) << 3;    // rows 0..7 (lanes 0-15) or 8..15 (16-31)
#pragma unroll
    for (int s = 0; s < 4; ++s) {
        const int n = wv * 64 + s * 16 + (lane & 15);
        const float bn = bias[n];
#pragma unroll
        for (int j = 0; j < 8; ++j) {
            const int m = mbase + rofs + j;
            Hout[(size_t)m * PROJ + n] = acc[s][j] + bn;
        }
    }
}

// Kernel: fused V-GEMM + tanh(H+V)@w_attn, z written in reference "scrambled" order.
__global__ __launch_bounds__(256) void k_attn_main(const float* __restrict__ F,
                                                   const bf16_t* __restrict__ Wbf,
                                                   const float* __restrict__ bcnn,
                                                   const float* __restrict__ Hbuf,
                                                   const float* __restrict__ wattn,
                                                   float* __restrict__ zbuf) {
    __shared__ float zsh[16];
    if (threadIdx.x < 16) zsh[threadIdx.x] = 0.0f;

    v8f acc[4];
#pragma unroll
    for (int s = 0; s < 4; ++s)
#pragma unroll
        for (int j = 0; j < 8; ++j) acc[s][j] = 0.0f;

    const int mbase = blockIdx.x * 16;
    gemm16xN<CNN_C, CNN_C>(F, Wbf, mbase, acc);

    const int lane = threadIdx.x & 31;
    const int wv   = threadIdx.x >> 5;
    const int rofs = (lane >> 4) << 3;

    float zp[8];
#pragma unroll
    for (int j = 0; j < 8; ++j) zp[j] = 0.0f;

#pragma unroll
    for (int s = 0; s < 4; ++s) {
        const int n  = wv * 64 + s * 16 + (lane & 15);
        const float bn = bcnn[n];
        const float wa = wattn[n];
#pragma unroll
        for (int j = 0; j < 8; ++j) {
            const int m = mbase + rofs + j;
            const int b = m / GRID_N;
            const float v = acc[s][j] + bn + Hbuf[(size_t)b * PROJ + n];
            zp[j] += tanhf(v) * wa;
        }
    }
    // reduce over the 16 lane-columns inside each 16-lane half
#pragma unroll
    for (int j = 0; j < 8; ++j)
#pragma unroll
        for (int off = 8; off >= 1; off >>= 1)
            zp[j] += __shfl_xor(zp[j], off, 16);

    __syncthreads();                        // zsh init happens-before atomics
    if ((lane & 15) == 0) {
#pragma unroll
        for (int j = 0; j < 8; ++j)
            atomicAdd(&zsh[rofs + j], zp[j]);   // combine 8 waves (ds_add_f32)
    }
    __syncthreads();

    if (threadIdx.x < 16) {
        const int m = mbase + threadIdx.x;
        const int b = m / GRID_N;
        const int g = m - b * GRID_N;
        zbuf[g * BS + b] = zsh[threadIdx.x];    // torch cat-over-grids flatten order
    }
}

// Softmax over each row of the scrambled (256 x 49) view.
__global__ void k_softmax(const float* __restrict__ zbuf, float* __restrict__ out) {
    const int i = blockIdx.x * blockDim.x + threadIdx.x;
    if (i >= BS) return;
    const float* row = zbuf + i * GRID_N;
    float mx = -1e30f;
    for (int j = 0; j < GRID_N; ++j) mx = fmaxf(mx, row[j]);
    float sum = 0.0f;
    for (int j = 0; j < GRID_N; ++j) sum += __expf(row[j] - mx);
    const float inv = 1.0f / sum;
    for (int j = 0; j < GRID_N; ++j) out[i * GRID_N + j] = __expf(row[j] - mx) * inv;
}

extern "C" void kernel_launch(void* const* d_in, const int* in_sizes, int n_in,
                              void* d_out, int out_size, void* d_ws, size_t ws_size,
                              hipStream_t stream) {
    const float* lstm_hidden = (const float*)d_in[0];
    const float* cnn_feat    = (const float*)d_in[1];
    const float* W_lstm      = (const float*)d_in[2];
    const float* b_lstm      = (const float*)d_in[3];
    const float* W_cnn       = (const float*)d_in[4];
    const float* b_cnn       = (const float*)d_in[5];
    const float* w_attn      = (const float*)d_in[6];

    // Workspace layout (~3.06 MB total):
    char* ws = (char*)d_ws;
    float*  Hbuf     = (float*)ws;                         ws += (size_t)BS * PROJ * 4;      // 512 KB
    float*  zbuf     = (float*)ws;                         ws += (size_t)M_ROWS * 4;         // 50 KB
    bf16_t* Wcnn_bf  = (bf16_t*)ws;                        ws += (size_t)PROJ * CNN_C * 2;   // 2 MB
    bf16_t* Wlstm_bf = (bf16_t*)ws;                        /* + 512 KB */

    // One-shot weight conversions (fp32 -> bf16, RNE)
    k_f32_to_bf16<<<512, 256, 0, stream>>>(W_cnn,  (unsigned int*)Wcnn_bf,  PROJ * CNN_C / 4);
    k_f32_to_bf16<<<256, 256, 0, stream>>>(W_lstm, (unsigned int*)Wlstm_bf, PROJ * LSTM_HD / 4);

    k_proj_lstm<<<BS / 16,     256, 0, stream>>>(lstm_hidden, Wlstm_bf, b_lstm, Hbuf);
    k_attn_main<<<M_ROWS / 16, 256, 0, stream>>>(cnn_feat, Wcnn_bf, b_cnn, Hbuf, w_attn, zbuf);
    k_softmax  <<<(BS + 63) / 64, 64, 0, stream>>>(zbuf, (float*)d_out);
}
